// FEDformerEncoderLayer_70411693850730
// MI455X (gfx1250) — matrix operand order repaired and checked by reading, not compiled, not run on hardware
//
#include <hip/hip_runtime.h>
#include <hip/hip_bf16.h>
#include <math.h>

// FEDformer encoder layer constants
#define Bsz   16
#define Ssz   4096
#define Dsz   512
#define Hsz   8
#define HDsz  64
#define DFFsz 2048
#define Ksel  64
#define NF    2049      // rfft bins of 4096
#define Mrows (Bsz*Ssz) // 65536

typedef __attribute__((ext_vector_type(16))) __bf16 v16bf;
typedef __attribute__((ext_vector_type(8)))  __bf16 v8bf;
typedef __attribute__((ext_vector_type(8)))  float  v8f;

__device__ __forceinline__ __bf16 f2bf(float f) {
  union { float f; unsigned u; } x; x.f = f;
  unsigned r = x.u + 0x7FFFu + ((x.u >> 16) & 1u);   // round-to-nearest-even
  unsigned short h = (unsigned short)(r >> 16);
  __bf16 b; __builtin_memcpy(&b, &h, sizeof(b));
  return b;
}

// A-fragment: 16-bit A 16x32 layout — lanes 0-15 hold K={0..7,16..23},
// lanes 16-31 hold K={8..15,24..31} (two 8-element chunks per lane).
__device__ __forceinline__ v16bf load_a_frag(const __bf16* row, int kh8) {
  v8bf lo = *(const v8bf*)(row + kh8);
  v8bf hi = *(const v8bf*)(row + kh8 + 16);
  v16bf r;
#pragma unroll
  for (int i = 0; i < 8; ++i) { r[i] = lo[i]; r[i + 8] = hi[i]; }
  return r;
}

// ---------------------------------------------------------------------------
// One-shot fp32 -> bf16 convert (8 elements / thread, b128 in, b128 out)
// ---------------------------------------------------------------------------
__global__ __launch_bounds__(256)
void convert_bf16_kernel(const float* __restrict__ in, __bf16* __restrict__ out)
{
  size_t t = ((size_t)blockIdx.x * 256 + threadIdx.x) * 8;
  float4 a = *(const float4*)(in + t);
  float4 b = *(const float4*)(in + t + 4);
  union { __bf16 h[8]; uint4 u; } r;
  r.h[0] = f2bf(a.x); r.h[1] = f2bf(a.y); r.h[2] = f2bf(a.z); r.h[3] = f2bf(a.w);
  r.h[4] = f2bf(b.x); r.h[5] = f2bf(b.y); r.h[6] = f2bf(b.z); r.h[7] = f2bf(b.w);
  *(uint4*)(out + t) = r.u;
}

// ---------------------------------------------------------------------------
// One-shot weight transpose + convert: W[k][n] (fp32, KxN) -> Wt[n][k] (bf16)
// ---------------------------------------------------------------------------
__global__ __launch_bounds__(256)
void transpose_bf16_kernel(const float* __restrict__ W, __bf16* __restrict__ Wt,
                           int K, int N)
{
  __shared__ float tile[32][33];
  int tx = threadIdx.x & 31;
  int ty = threadIdx.x >> 5;            // 0..7
  int nb = blockIdx.x * 32;             // n tile
  int kb = blockIdx.y * 32;             // k tile
#pragma unroll
  for (int j = 0; j < 32; j += 8)
    tile[ty + j][tx] = W[(size_t)(kb + ty + j) * N + (nb + tx)];
  __syncthreads();
#pragma unroll
  for (int j = 0; j < 32; j += 8)
    Wt[(size_t)(nb + ty + j) * K + (kb + tx)] = f2bf(tile[tx][ty + j]);
}

// ---------------------------------------------------------------------------
// GEMM: C = act(A(bf16,MxK) * Bt(bf16,NxK)^T + bias) [+ resid(fp32)]
// Block tile 128x128, BK=64. 8 waves as 4(M)x2(N); each wave computes 32x64
// via 2x4 16x16 WMMA tiles (16 v_wmma per staged tile). Register
// double-buffering with scalarized uint4 temps; __launch_bounds__(256,1)
// releases the VGPR budget so the staging buffer stays in registers.
// ---------------------------------------------------------------------------
__global__ __launch_bounds__(256, 1)
void gemm_bf16_wmma(const __bf16* __restrict__ A, const __bf16* __restrict__ Bt,
                    const float* __restrict__ bias, const float* __restrict__ resid,
                    float* __restrict__ Cf, __bf16* __restrict__ Cbf,
                    int N, int Kd, int relu)
{
  __shared__ __bf16 sA[128 * 64];       // [m][k] tile
  __shared__ __bf16 sB[128 * 64];       // [n][k] tile (Bt already transposed)

  const int tid   = threadIdx.x;
  const int lane  = tid & 31;
  const int wave  = tid >> 5;
  const int waveM = wave >> 1;          // 0..3
  const int waveN = wave & 1;           // 0..1
  const int m0 = blockIdx.y * 128;
  const int n0 = blockIdx.x * 128;

  const int lr   = lane & 15;           // row (A) / col (B) within 16-tile
  const int kh8  = (lane >> 4) * 8;     // table-order K chunk select (unused)
  const int kh16 = (lane >> 4) * 16;    // B-fragment K half select

  // Per-thread staging geometry: 4 chunks of 8 bf16, strided by 256 threads
  const int q0 = tid * 8;
  const int r0 = q0 >> 6,          c0 = q0 & 63;
  const int r1 = (q0 + 2048) >> 6, c1 = c0;   // q + 256*8
  const int r2 = (q0 + 4096) >> 6, c2 = c0;
  const int r3 = (q0 + 6144) >> 6, c3 = c0;

  const __bf16* pa = &A[(size_t)(m0 + r0) * Kd + c0];
  const __bf16* pb = &Bt[(size_t)(n0 + r0) * Kd + c0];
  const size_t rowstep = (size_t)32 * Kd;     // 2048/64 rows * Kd

  v8f acc[2][4] = {};

  // Prologue: first K-tile -> scalarized register double-buffer
  uint4 ra0 = *(const uint4*)(pa);
  uint4 ra1 = *(const uint4*)(pa + rowstep);
  uint4 ra2 = *(const uint4*)(pa + 2 * rowstep);
  uint4 ra3 = *(const uint4*)(pa + 3 * rowstep);
  uint4 rb0 = *(const uint4*)(pb);
  uint4 rb1 = *(const uint4*)(pb + rowstep);
  uint4 rb2 = *(const uint4*)(pb + 2 * rowstep);
  uint4 rb3 = *(const uint4*)(pb + 3 * rowstep);

  for (int k0 = 0; k0 < Kd; k0 += 64) {
    // Dump staged registers to LDS
    *(uint4*)&sA[r0 * 64 + c0] = ra0;
    *(uint4*)&sA[r1 * 64 + c1] = ra1;
    *(uint4*)&sA[r2 * 64 + c2] = ra2;
    *(uint4*)&sA[r3 * 64 + c3] = ra3;
    *(uint4*)&sB[r0 * 64 + c0] = rb0;
    *(uint4*)&sB[r1 * 64 + c1] = rb1;
    *(uint4*)&sB[r2 * 64 + c2] = rb2;
    *(uint4*)&sB[r3 * 64 + c3] = rb3;
    __syncthreads();

    // Issue next K-tile's global loads (latency hidden behind WMMAs)
    if (k0 + 64 < Kd) {
      const __bf16* na = pa + (k0 + 64);
      const __bf16* nb = pb + (k0 + 64);
      ra0 = *(const uint4*)(na);
      ra1 = *(const uint4*)(na + rowstep);
      ra2 = *(const uint4*)(na + 2 * rowstep);
      ra3 = *(const uint4*)(na + 3 * rowstep);
      rb0 = *(const uint4*)(nb);
      rb1 = *(const uint4*)(nb + rowstep);
      rb2 = *(const uint4*)(nb + 2 * rowstep);
      rb3 = *(const uint4*)(nb + 3 * rowstep);
    }

    // 16 WMMAs over the staged tile (2 K-steps x 2 M-tiles x 4 N-tiles)
#pragma unroll
    for (int ks = 0; ks < 64; ks += 32) {
      v16bf a0 = *(const v16bf*)&sA[(waveM * 32 +  0 + lr) * 64 + ks + kh16];
      v16bf a1 = *(const v16bf*)&sA[(waveM * 32 + 16 + lr) * 64 + ks + kh16];
#pragma unroll
      for (int tn = 0; tn < 4; ++tn) {
        v16bf b = *(const v16bf*)&sB[(waveN * 64 + tn * 16 + lr) * 64 + ks + kh16];
        acc[0][tn] = __builtin_amdgcn_wmma_f32_16x16x32_bf16(false, a0, false, b,
                       (short)0, acc[0][tn], false, false);
        acc[1][tn] = __builtin_amdgcn_wmma_f32_16x16x32_bf16(false, a1, false, b,
                       (short)0, acc[1][tn], false, false);
      }
    }
    __syncthreads();
  }

  // Epilogue: C/D 16x16 f32 layout — VGPR r, lanes 0-15 -> M=r, 16-31 -> M=8+r
  const int rbase = (lane >> 4) * 8;
#pragma unroll
  for (int tm = 0; tm < 2; ++tm) {
#pragma unroll
    for (int tn = 0; tn < 4; ++tn) {
      int colg = n0 + waveN * 64 + tn * 16 + lr;
      float bcol = bias[colg];
#pragma unroll
      for (int r = 0; r < 8; ++r) {
        int rowg = m0 + waveM * 32 + tm * 16 + rbase + r;
        float val = acc[tm][tn][r] + bcol;
        if (relu) val = fmaxf(val, 0.0f);
        size_t off = (size_t)rowg * N + colg;
        if (resid) val += resid[off];
        if (Cbf) Cbf[off] = f2bf(val);
        else     Cf[off]  = val;
      }
    }
  }
}

// ---------------------------------------------------------------------------
// 4096-point radix-2 FFT core in LDS (bit-reversed DIT), 256 threads.
// ---------------------------------------------------------------------------
__device__ __forceinline__ int bitrev12(int x) {
  return (int)(__brev((unsigned)x) >> 20);
}

__device__ void fft4096(float* re, float* im, int tid) {
  for (int stage = 0; stage < 12; ++stage) {
    int half = 1 << stage;
    for (int i = tid; i < 2048; i += 256) {
      int pos  = i & (half - 1);
      int idx0 = ((i >> stage) << (stage + 1)) + pos;
      int idx1 = idx0 + half;
      float ang = -3.14159265358979323846f * (float)pos / (float)half;
      float s, c; __sincosf(ang, &s, &c);
      float br = re[idx1], bi = im[idx1];
      float xr = br * c - bi * s;
      float xi = br * s + bi * c;
      float ar = re[idx0], ai = im[idx0];
      re[idx1] = ar - xr; im[idx1] = ai - xi;
      re[idx0] = ar + xr; im[idx0] = ai + xi;
    }
    __syncthreads();
  }
}

// rfft along S for one (b,h,d) column of a (B,S,D) tensor -> (B,H,NF,HD) cplx
__global__ __launch_bounds__(256)
void rfft_kernel(const float* __restrict__ in, float2* __restrict__ outf)
{
  __shared__ float re[4096], im[4096];
  int col = blockIdx.x;                  // b*512 + h*64 + d
  int d = col & 63, h = (col >> 6) & 7, b = col >> 9;
  int tid = threadIdx.x;
  size_t inb = (size_t)b * Ssz * Dsz + (h * HDsz + d);
  for (int s = tid; s < 4096; s += 256) {
    int r = bitrev12(s);
    re[r] = in[inb + (size_t)s * Dsz];
    im[r] = 0.0f;
  }
  __syncthreads();
  fft4096(re, im, tid);
  size_t ob = (size_t)(b * Hsz + h) * NF * HDsz + d;
  for (int f = tid; f < NF; f += 256) {
    float2 v; v.x = re[f]; v.y = im[f];
    outf[ob + (size_t)f * HDsz] = v;
  }
}

// irfft: rebuild Hermitian spectrum, inverse = conj(FFT(conj(X)))/N, keep real.
// Writes bf16 directly (consumer is the Wo WMMA GEMM).
__global__ __launch_bounds__(256)
void irfft_kernel(const float2* __restrict__ spec, __bf16* __restrict__ out)
{
  __shared__ float re[4096], im[4096];
  int col = blockIdx.x;
  int d = col & 63, h = (col >> 6) & 7, b = col >> 9;
  int tid = threadIdx.x;
  size_t sb = (size_t)(b * Hsz + h) * NF * HDsz + d;
  for (int s = tid; s < 4096; s += 256) {
    float2 c;
    if (s <= 2048) c = spec[sb + (size_t)s * HDsz];
    else { float2 t = spec[sb + (size_t)(4096 - s) * HDsz]; c.x = t.x; c.y = -t.y; }
    int r = bitrev12(s);
    re[r] = c.x; im[r] = -c.y;           // conjugate for inverse transform
  }
  __syncthreads();
  fft4096(re, im, tid);
  size_t ob = (size_t)b * Ssz * Dsz + (h * HDsz + d);
  const float inv = 1.0f / 4096.0f;
  for (int s = tid; s < 4096; s += 256)
    out[ob + (size_t)s * Dsz] = f2bf(re[s] * inv);
}

// ---------------------------------------------------------------------------
// Top-64 frequencies by mean |xf| over HD, per (b,h); iterative argmax.
// ---------------------------------------------------------------------------
__global__ __launch_bounds__(256)
void topk_kernel(const float2* __restrict__ xf, int* __restrict__ idx)
{
  __shared__ float mag[NF];
  __shared__ float rv[256];
  __shared__ int   ri[256];
  int bh = blockIdx.x;
  int tid = threadIdx.x;
  size_t base = (size_t)bh * NF * HDsz;
  for (int f = tid; f < NF; f += 256) {
    float s = 0.0f;
    const float2* p = xf + base + (size_t)f * HDsz;
    for (int d2 = 0; d2 < HDsz; ++d2) {
      float2 c = p[d2];
      s += sqrtf(c.x * c.x + c.y * c.y);
    }
    mag[f] = s;
  }
  __syncthreads();
  for (int j = 0; j < Ksel; ++j) {
    float best = -1.0f; int bi = NF;
    for (int f = tid; f < NF; f += 256)
      if (mag[f] > best) { best = mag[f]; bi = f; }
    rv[tid] = best; ri[tid] = bi;
    __syncthreads();
    for (int off = 128; off; off >>= 1) {
      if (tid < off) {
        if (rv[tid + off] > rv[tid] ||
            (rv[tid + off] == rv[tid] && ri[tid + off] < ri[tid])) {
          rv[tid] = rv[tid + off]; ri[tid] = ri[tid + off];
        }
      }
      __syncthreads();
    }
    if (tid == 0) { idx[bh * Ksel + j] = ri[0]; mag[ri[0]] = -3.0e38f; }
    __syncthreads();
  }
}

__global__ void zero_kernel(float* p, size_t n) {
  size_t t = (size_t)blockIdx.x * 256 + threadIdx.x;
  if (t < n) p[t] = 0.0f;
}

// out_f = q_sel * conj(k_sel) * v_sel * (fw0 + i fw1), scattered at q_idx
__global__ __launch_bounds__(256)
void combine_kernel(const float2* __restrict__ qf, const float2* __restrict__ kf,
                    const float2* __restrict__ vf, const int* __restrict__ iq,
                    const int* __restrict__ ik, const int* __restrict__ iv,
                    const float* __restrict__ fw, float2* __restrict__ spec)
{
  int t = blockIdx.x * 256 + threadIdx.x;   // (b,h,j,d)
  int d = t & 63, j = (t >> 6) & 63, h = (t >> 12) & 7, b = t >> 15;
  int bh = b * Hsz + h;
  int qi = iq[bh * Ksel + j];
  int ki = ik[bh * Ksel + j];
  int vi = iv[bh * Ksel + j];
  size_t base = (size_t)bh * NF * HDsz;
  float2 q = qf[base + (size_t)qi * HDsz + d];
  float2 k = kf[base + (size_t)ki * HDsz + d];
  float2 v = vf[base + (size_t)vi * HDsz + d];
  float pr = q.x * k.x + q.y * k.y;         // q * conj(k)
  float pi = q.y * k.x - q.x * k.y;
  float r2 = pr * v.x - pi * v.y;           // * v
  float i2 = pr * v.y + pi * v.x;
  float wr = fw[((size_t)(0 * Hsz + h) * Ksel + j) * HDsz + d];
  float wi = fw[((size_t)(1 * Hsz + h) * Ksel + j) * HDsz + d];
  float2 o; o.x = r2 * wr - i2 * wi; o.y = r2 * wi + i2 * wr;
  spec[base + (size_t)qi * HDsz + d] = o;
}

// out = in - moving_avg(in, 25), edge (clamp) padding along S.
// Optionally dual-writes a bf16 copy (for the next WMMA GEMM's A operand).
__global__ __launch_bounds__(256)
void mavg_sub_kernel(const float* __restrict__ in, float* __restrict__ outf,
                     __bf16* __restrict__ outb)
{
  size_t t = (size_t)blockIdx.x * 256 + threadIdx.x;
  int d = (int)(t % Dsz);
  size_t sb = t / Dsz;
  int s = (int)(sb % Ssz);
  int b = (int)(sb / Ssz);
  size_t rowb = (size_t)b * Ssz * Dsz + d;
  float sum = 0.0f;
#pragma unroll
  for (int j = -12; j <= 12; ++j) {
    int ss = s + j;
    ss = ss < 0 ? 0 : (ss > Ssz - 1 ? Ssz - 1 : ss);
    sum += in[rowb + (size_t)ss * Dsz];
  }
  float v = in[t] - sum * (1.0f / 25.0f);
  outf[t] = v;
  if (outb) outb[t] = f2bf(v);
}

// ---------------------------------------------------------------------------
extern "C" void kernel_launch(void* const* d_in, const int* in_sizes, int n_in,
                              void* d_out, int out_size, void* d_ws, size_t ws_size,
                              hipStream_t stream)
{
  const float* x  = (const float*)d_in[0];
  const float* Wq = (const float*)d_in[1];
  const float* bq = (const float*)d_in[2];
  const float* Wk = (const float*)d_in[3];
  const float* bk = (const float*)d_in[4];
  const float* Wv = (const float*)d_in[5];
  const float* bv = (const float*)d_in[6];
  const float* Wo = (const float*)d_in[7];
  const float* bo = (const float*)d_in[8];
  const float* fw = (const float*)d_in[9];
  const float* W1 = (const float*)d_in[10];
  const float* b1 = (const float*)d_in[11];
  const float* W2 = (const float*)d_in[12];
  const float* b2 = (const float*)d_in[13];
  float* out = (float*)d_out;

  char* ws = (char*)d_ws;
  const size_t SZ_BSD  = (size_t)Mrows * Dsz * sizeof(float);            // 128 MB
  const size_t SZ_BSDH = (size_t)Mrows * Dsz * sizeof(__bf16);           //  64 MB
  const size_t SZ_F    = (size_t)Bsz * Hsz * NF * HDsz * sizeof(float2); // ~128 MB
  const size_t SZ_IDX  = (size_t)Bsz * Hsz * Ksel * sizeof(int);
  const size_t SZ_WDD  = (size_t)Dsz * Dsz * sizeof(__bf16);
  const size_t SZ_WDF  = (size_t)Dsz * DFFsz * sizeof(__bf16);

  size_t o = 0;
  __bf16* xbf    = (__bf16*)(ws + o); o += SZ_BSDH;
  float*  qbuf   = (float*)(ws + o);  o += SZ_BSD;
  float*  kbuf   = (float*)(ws + o);  o += SZ_BSD;
  float*  vbuf   = (float*)(ws + o);  o += SZ_BSD;
  float2* qf     = (float2*)(ws + o); o += SZ_F;
  float2* kf     = (float2*)(ws + o); o += SZ_F;
  float2* vf     = (float2*)(ws + o); o += SZ_F;
  float2* spec   = (float2*)(ws + o); o += SZ_F;
  int*    iq     = (int*)(ws + o);    o += SZ_IDX;
  int*    ik     = (int*)(ws + o);    o += SZ_IDX;
  int*    iv     = (int*)(ws + o);    o += SZ_IDX;
  __bf16* attnbf = (__bf16*)(ws + o); o += SZ_BSDH;
  __bf16* x1sbf  = (__bf16*)(ws + o); o += SZ_BSDH;
  __bf16* hidden = (__bf16*)(ws + o); o += (size_t)Mrows * DFFsz * sizeof(__bf16);
  __bf16* Wqt    = (__bf16*)(ws + o); o += SZ_WDD;
  __bf16* Wkt    = (__bf16*)(ws + o); o += SZ_WDD;
  __bf16* Wvt    = (__bf16*)(ws + o); o += SZ_WDD;
  __bf16* Wot    = (__bf16*)(ws + o); o += SZ_WDD;
  __bf16* W1t    = (__bf16*)(ws + o); o += SZ_WDF;
  __bf16* W2t    = (__bf16*)(ws + o); o += SZ_WDF;
  // fp32 buffer reuse after the rfft stage consumes qbuf/kbuf/vbuf:
  float* x1   = qbuf;   // x + attn@Wo + bo
  float* x1sf = kbuf;   // seasonal after decomp1
  float* z    = vbuf;   // FFN output + residual

  dim3 blk(256);
  dim3 gD(Dsz / 128, Mrows / 128);    // (4, 512)
  dim3 gF(DFFsz / 128, Mrows / 128);  // (16, 512)
  const int ncols = Bsz * Hsz * HDsz; // 8192 FFT columns
  size_t nBSD = (size_t)Mrows * Dsz;

  // One-shot precision/layout prep
  convert_bf16_kernel<<<(unsigned)(nBSD / (256 * 8)), blk, 0, stream>>>(x, xbf);
  transpose_bf16_kernel<<<dim3(Dsz/32, Dsz/32), blk, 0, stream>>>(Wq, Wqt, Dsz, Dsz);
  transpose_bf16_kernel<<<dim3(Dsz/32, Dsz/32), blk, 0, stream>>>(Wk, Wkt, Dsz, Dsz);
  transpose_bf16_kernel<<<dim3(Dsz/32, Dsz/32), blk, 0, stream>>>(Wv, Wvt, Dsz, Dsz);
  transpose_bf16_kernel<<<dim3(Dsz/32, Dsz/32), blk, 0, stream>>>(Wo, Wot, Dsz, Dsz);
  transpose_bf16_kernel<<<dim3(DFFsz/32, Dsz/32), blk, 0, stream>>>(W1, W1t, Dsz, DFFsz);
  transpose_bf16_kernel<<<dim3(Dsz/32, DFFsz/32), blk, 0, stream>>>(W2, W2t, DFFsz, Dsz);

  // Q/K/V projections (bf16 WMMA, fused bias)
  gemm_bf16_wmma<<<gD, blk, 0, stream>>>(xbf, Wqt, bq, nullptr, qbuf, nullptr, Dsz, Dsz, 0);
  gemm_bf16_wmma<<<gD, blk, 0, stream>>>(xbf, Wkt, bk, nullptr, kbuf, nullptr, Dsz, Dsz, 0);
  gemm_bf16_wmma<<<gD, blk, 0, stream>>>(xbf, Wvt, bv, nullptr, vbuf, nullptr, Dsz, Dsz, 0);

  // rfft along S
  rfft_kernel<<<ncols, blk, 0, stream>>>(qbuf, qf);
  rfft_kernel<<<ncols, blk, 0, stream>>>(kbuf, kf);
  rfft_kernel<<<ncols, blk, 0, stream>>>(vbuf, vf);

  // top-64 frequency selection per (b,h)
  topk_kernel<<<Bsz * Hsz, blk, 0, stream>>>(qf, iq);
  topk_kernel<<<Bsz * Hsz, blk, 0, stream>>>(kf, ik);
  topk_kernel<<<Bsz * Hsz, blk, 0, stream>>>(vf, iv);

  // zero spectrum, combine + scatter at q_idx
  size_t specN = (size_t)Bsz * Hsz * NF * HDsz * 2;
  zero_kernel<<<(unsigned)((specN + 255) / 256), blk, 0, stream>>>((float*)spec, specN);
  combine_kernel<<<(Bsz * Hsz * Ksel * HDsz) / 256, blk, 0, stream>>>(
      qf, kf, vf, iq, ik, iv, fw, spec);

  // irfft -> attention heads back to (B,S,D), bf16 for the Wo GEMM
  irfft_kernel<<<ncols, blk, 0, stream>>>(spec, attnbf);

  // output projection + residual x -> x1 (fp32)
  gemm_bf16_wmma<<<gD, blk, 0, stream>>>(attnbf, Wot, bo, x, x1, nullptr, Dsz, Dsz, 0);

  // decomp1: x1s = x1 - mavg(x1); dual write fp32 + bf16
  mavg_sub_kernel<<<(unsigned)(nBSD / 256), blk, 0, stream>>>(x1, x1sf, x1sbf);

  // FFN: hidden = relu(x1s @ W1 + b1) (bf16 out); z = hidden @ W2 + b2 + x1s
  gemm_bf16_wmma<<<gF, blk, 0, stream>>>(x1sbf, W1t, b1, nullptr, nullptr, hidden,
                                         DFFsz, Dsz, 1);
  gemm_bf16_wmma<<<gD, blk, 0, stream>>>(hidden, W2t, b2, x1sf, z, nullptr,
                                         Dsz, DFFsz, 0);

  // decomp2 -> output
  mavg_sub_kernel<<<(unsigned)(nBSD / 256), blk, 0, stream>>>(z, out, nullptr);
}
